// IMP_56788057588234
// MI455X (gfx1250) — matrix operand order repaired and checked by reading, not compiled, not run on hardware
//
#include <hip/hip_runtime.h>

#define N_OBJ  16384
#define N_PAIR 131072
#define DD     256
#define D3     768

typedef __attribute__((ext_vector_type(16))) __bf16 v16bf;
typedef __attribute__((ext_vector_type(8)))  __bf16 v8bf;
typedef __attribute__((ext_vector_type(8)))  float  v8f;

__device__ __forceinline__ unsigned short f2bf_bits(float f) {
    unsigned u = __builtin_bit_cast(unsigned, f);
    u += 0x7FFFu + ((u >> 16) & 1u);          // round-to-nearest-even
    return (unsigned short)(u >> 16);
}
__device__ __forceinline__ float sigmoidf_(float x) { return 1.0f / (1.0f + __expf(-x)); }

__device__ __forceinline__ void atomAddF(float* p, float v) {
    unsafeAtomicAdd(p, v);                    // global_atomic_add_f32
}

// ---------------------------------------------------------------- zero
__global__ void k_zero(float* __restrict__ p, int n) {
    int i = blockIdx.x * blockDim.x + threadIdx.x;
    if (i < n) p[i] = 0.0f;
}

// ------------------------------------------------- weights f32 -> bf16
__global__ void k_cvt_w(const float* __restrict__ wih, const float* __restrict__ whh,
                        unsigned short* __restrict__ wihB, unsigned short* __restrict__ whhB) {
    int i = blockIdx.x * blockDim.x + threadIdx.x;
    if (i < D3 * DD) { wihB[i] = f2bf_bits(wih[i]); whhB[i] = f2bf_bits(whh[i]); }
}

// ---------------------------------------------------------- pair gates
// One wave32 per pair; lane owns 8 contiguous columns.
__global__ __launch_bounds__(256) void k_pair(
    const float* __restrict__ xobj, const float* __restrict__ xpred,
    const int* __restrict__ idx,
    const float* __restrict__ w_e2v, const float* __restrict__ b_e2v,
    const float* __restrict__ w_v2e, const float* __restrict__ b_v2e,
    float* __restrict__ msg_obj, float* __restrict__ pred_msg)
{
    const int lane   = threadIdx.x & 31;
    const int wave   = (blockIdx.x * blockDim.x + threadIdx.x) >> 5;
    const int nwaves = (gridDim.x * blockDim.x) >> 5;
    const int k      = lane * 8;

    const float4 wel0 = ((const float4*)(w_e2v + k))[0];
    const float4 wel1 = ((const float4*)(w_e2v + k))[1];
    const float4 weh0 = ((const float4*)(w_e2v + 256 + k))[0];
    const float4 weh1 = ((const float4*)(w_e2v + 256 + k))[1];
    const float4 wvl0 = ((const float4*)(w_v2e + k))[0];
    const float4 wvl1 = ((const float4*)(w_v2e + k))[1];
    const float4 wvh0 = ((const float4*)(w_v2e + 256 + k))[0];
    const float4 wvh1 = ((const float4*)(w_v2e + 256 + k))[1];
    const float  be   = b_e2v[0], bv = b_v2e[0];

    for (int p = wave; p < N_PAIR; p += nwaves) {
        const int s = idx[2 * p + 0];
        const int o = idx[2 * p + 1];
        const float4* xp = (const float4*)(xpred + (size_t)p * DD + k);
        const float4* xs = (const float4*)(xobj  + (size_t)s * DD + k);
        const float4* xo = (const float4*)(xobj  + (size_t)o * DD + k);
        float4 p0 = xp[0], p1 = xp[1];
        float4 s0 = xs[0], s1 = xs[1];
        float4 o0 = xo[0], o1 = xo[1];

        float d_se = 0.f, d_pe = 0.f, d_oe = 0.f, d_pv = 0.f, d_sv = 0.f, d_ov = 0.f;
#define ACC1(PX, SX, OX, WEL, WEH, WVL, WVH)                         \
        { float rp = fmaxf((PX), 0.f), rs = fmaxf((SX), 0.f),        \
                ro = fmaxf((OX), 0.f);                               \
          d_se += rs * (WEL); d_pe += rp * (WEH); d_oe += ro * (WEL);\
          d_pv += rp * (WVL); d_sv += rs * (WVH); d_ov += ro * (WVH); }
        ACC1(p0.x, s0.x, o0.x, wel0.x, weh0.x, wvl0.x, wvh0.x);
        ACC1(p0.y, s0.y, o0.y, wel0.y, weh0.y, wvl0.y, wvh0.y);
        ACC1(p0.z, s0.z, o0.z, wel0.z, weh0.z, wvl0.z, wvh0.z);
        ACC1(p0.w, s0.w, o0.w, wel0.w, weh0.w, wvl0.w, wvh0.w);
        ACC1(p1.x, s1.x, o1.x, wel1.x, weh1.x, wvl1.x, wvh1.x);
        ACC1(p1.y, s1.y, o1.y, wel1.y, weh1.y, wvl1.y, wvh1.y);
        ACC1(p1.z, s1.z, o1.z, wel1.z, weh1.z, wvl1.z, wvh1.z);
        ACC1(p1.w, s1.w, o1.w, wel1.w, weh1.w, wvl1.w, wvh1.w);
#undef ACC1
        #pragma unroll
        for (int off = 16; off > 0; off >>= 1) {
            d_se += __shfl_xor(d_se, off, 32);
            d_pe += __shfl_xor(d_pe, off, 32);
            d_oe += __shfl_xor(d_oe, off, 32);
            d_pv += __shfl_xor(d_pv, off, 32);
            d_sv += __shfl_xor(d_sv, off, 32);
            d_ov += __shfl_xor(d_ov, off, 32);
        }
        const float gs  = sigmoidf_(d_se + d_pe + be);   // gate for m_subj
        const float go  = sigmoidf_(d_oe + d_pe + be);   // gate for m_obj
        const float gsp = sigmoidf_(d_pv + d_sv + bv);   // gate for m_s2p
        const float gop = sigmoidf_(d_pv + d_ov + bv);   // gate for m_o2p

        float4 m0, m1;
        m0.x = s0.x * gsp + o0.x * gop;  m0.y = s0.y * gsp + o0.y * gop;
        m0.z = s0.z * gsp + o0.z * gop;  m0.w = s0.w * gsp + o0.w * gop;
        m1.x = s1.x * gsp + o1.x * gop;  m1.y = s1.y * gsp + o1.y * gop;
        m1.z = s1.z * gsp + o1.z * gop;  m1.w = s1.w * gsp + o1.w * gop;
        ((float4*)(pred_msg + (size_t)p * DD + k))[0] = m0;
        ((float4*)(pred_msg + (size_t)p * DD + k))[1] = m1;

        float* ds = msg_obj + (size_t)s * DD + k;
        float* dd = msg_obj + (size_t)o * DD + k;
        atomAddF(ds + 0, p0.x * gs); atomAddF(ds + 1, p0.y * gs);
        atomAddF(ds + 2, p0.z * gs); atomAddF(ds + 3, p0.w * gs);
        atomAddF(ds + 4, p1.x * gs); atomAddF(ds + 5, p1.y * gs);
        atomAddF(ds + 6, p1.z * gs); atomAddF(ds + 7, p1.w * gs);
        atomAddF(dd + 0, p0.x * go); atomAddF(dd + 1, p0.y * go);
        atomAddF(dd + 2, p0.z * go); atomAddF(dd + 3, p0.w * go);
        atomAddF(dd + 4, p1.x * go); atomAddF(dd + 5, p1.y * go);
        atomAddF(dd + 6, p1.z * go); atomAddF(dd + 7, p1.w * go);
    }
}

// ------------------------------------------------------- fused GRU step
// Block = 32 rows x full 768 GEMM cols. 8 waves; wave w owns output cols
// [32w,32w+32): GEMM n-tiles {2w,2w+1} + {0,16,32} for both W_ih and W_hh,
// applied to 2 row-tiles (M=32) so each B fragment feeds 2 WMMAs.
// r/z/n combine happens in registers (identical C-tile layouts).
__global__ __launch_bounds__(256) void k_gru(
    const float* __restrict__ Ain,                 // M x 256 (message -> gi)
    const float* __restrict__ Hin,                 // M x 256 (hidden  -> gh)
    const unsigned short* __restrict__ WihB,       // 768 x 256 bf16, row-major
    const unsigned short* __restrict__ WhhB,       // 768 x 256 bf16, row-major
    const float* __restrict__ b_ih, const float* __restrict__ b_hh,
    float* __restrict__ Out, int M)
{
    (void)M;
    __shared__ __align__(16) unsigned short sA[32 * DD];   // bf16 A tile (16 KB)
    __shared__ __align__(16) unsigned short sH[32 * DD];   // bf16 H tile (16 KB)

    const int tid   = threadIdx.x;
    const int wave  = tid >> 5;
    const int lane  = tid & 31;
    const int mbase = blockIdx.x * 32;

    const float* Ag = Ain + (size_t)mbase * DD;
    const float* Hg = Hin + (size_t)mbase * DD;
    for (int i = tid; i < 32 * DD; i += 256) {
        sA[i] = f2bf_bits(Ag[i]);
        sH[i] = f2bf_bits(Hg[i]);
    }
    __syncthreads();

    const int nsel = (lane >> 4) & 1;   // half-wave select per ISA frag layout
    const int mrow = lane & 15;
    const int t0   = 2 * wave;

    v8f accI[2][6], accH[2][6];
    const v8f vzero = {0.f, 0.f, 0.f, 0.f, 0.f, 0.f, 0.f, 0.f};
    #pragma unroll
    for (int mt = 0; mt < 2; ++mt)
        #pragma unroll
        for (int j = 0; j < 6; ++j) { accI[mt][j] = vzero; accH[mt][j] = vzero; }

    #pragma unroll
    for (int kt = 0; kt < 8; ++kt) {
        const int k0 = kt * 32;
        // A-fragments (16x32 bf16): lanes 0-15 -> K {0..15}, 16-31 -> {16..31}
        v16bf aA[2], aH[2];
        #pragma unroll
        for (int mt = 0; mt < 2; ++mt) {
            const unsigned short* rA = sA + (mt * 16 + mrow) * DD + k0 + nsel * 16;
            const unsigned short* rH = sH + (mt * 16 + mrow) * DD + k0 + nsel * 16;
            v8bf alo = *(const v8bf*)(rA);
            v8bf ahi = *(const v8bf*)(rA + 8);
            v8bf hlo = *(const v8bf*)(rH);
            v8bf hhi = *(const v8bf*)(rH + 8);
            #pragma unroll
            for (int i = 0; i < 8; ++i) {
                aA[mt][i] = alo[i]; aA[mt][8 + i] = ahi[i];
                aH[mt][i] = hlo[i]; aH[mt][8 + i] = hhi[i];
            }
        }
        #pragma unroll
        for (int j = 0; j < 6; ++j) {
            const int tile = t0 + (j & 1) + (j >> 1) * 16;
            // B-fragment (32x16): lane-half owns 16 contiguous K of column row
            const size_t boff = (size_t)(tile * 16 + mrow) * DD + k0 + nsel * 16;
            v16bf bI = *(const v16bf*)(WihB + boff);
            v16bf bH = *(const v16bf*)(WhhB + boff);
            accI[0][j] = __builtin_amdgcn_wmma_f32_16x16x32_bf16(
                             false, aA[0], false, bI, (short)0, accI[0][j], false, false);
            accI[1][j] = __builtin_amdgcn_wmma_f32_16x16x32_bf16(
                             false, aA[1], false, bI, (short)0, accI[1][j], false, false);
            accH[0][j] = __builtin_amdgcn_wmma_f32_16x16x32_bf16(
                             false, aH[0], false, bH, (short)0, accH[0][j], false, false);
            accH[1][j] = __builtin_amdgcn_wmma_f32_16x16x32_bf16(
                             false, aH[1], false, bH, (short)0, accH[1][j], false, false);
        }
    }

    // Pointwise GRU combine in registers; C layout: VGPR e -> M=e (lanes 0-15),
    // M=8+e (lanes 16-31); N = lane&15 within the tile. Carry-path h is re-read
    // from global in f32 (L2-hot) to avoid bf16 rounding on the residual term.
    #pragma unroll
    for (int mt = 0; mt < 2; ++mt) {
        #pragma unroll
        for (int jj = 0; jj < 2; ++jj) {
            const int col = (t0 + jj) * 16 + mrow;     // 0..255 output column
            const float bir = b_ih[col],        bhr = b_hh[col];
            const float biz = b_ih[col + 256],  bhz = b_hh[col + 256];
            const float bin = b_ih[col + 512],  bhn = b_hh[col + 512];
            #pragma unroll
            for (int e = 0; e < 8; ++e) {
                const int row = mt * 16 + nsel * 8 + e;
                const float r  = sigmoidf_(accI[mt][jj][e]     + bir + accH[mt][jj][e]     + bhr);
                const float z  = sigmoidf_(accI[mt][2 + jj][e] + biz + accH[mt][2 + jj][e] + bhz);
                const float nn = tanhf(accI[mt][4 + jj][e] + bin + r * (accH[mt][4 + jj][e] + bhn));
                const float h  = Hg[(size_t)row * DD + col];
                Out[(size_t)(mbase + row) * DD + col] = (1.f - z) * nn + z * h;
            }
        }
    }
}

// ----------------------------------------------------------------------
extern "C" void kernel_launch(void* const* d_in, const int* in_sizes, int n_in,
                              void* d_out, int out_size, void* d_ws, size_t ws_size,
                              hipStream_t stream) {
    (void)in_sizes; (void)n_in; (void)out_size; (void)ws_size;
    const float* x_obj  = (const float*)d_in[0];
    const float* x_pred = (const float*)d_in[1];
    const int*   idx    = (const int*)  d_in[2];
    const float* w_e2v  = (const float*)d_in[3];
    const float* b_e2v  = (const float*)d_in[4];
    const float* w_v2e  = (const float*)d_in[5];
    const float* b_v2e  = (const float*)d_in[6];
    const float* w_ih   = (const float*)d_in[7];
    const float* w_hh   = (const float*)d_in[8];
    const float* b_ih   = (const float*)d_in[9];
    const float* b_hh   = (const float*)d_in[10];

    const size_t OBJ = (size_t)N_OBJ * DD;
    const size_t PRD = (size_t)N_PAIR * DD;
    float* ws        = (float*)d_ws;
    float* obj_A     = ws;                       // step-0 x_obj_new
    float* pred_A    = obj_A + OBJ;              // step-0 x_pred_new
    float* msg_obj   = pred_A + PRD;             // segment sums
    float* pred_msg  = msg_obj + OBJ;            // m_s2p + m_o2p
    unsigned short* wihB = (unsigned short*)(pred_msg + PRD);
    unsigned short* whhB = wihB + (size_t)D3 * DD;

    float* outObj  = (float*)d_out;
    float* outPred = outObj + OBJ;

    k_cvt_w<<<(D3 * DD + 255) / 256, 256, 0, stream>>>(w_ih, w_hh, wihB, whhB);

    const float* co = x_obj;
    const float* cp = x_pred;
    for (int step = 0; step < 2; ++step) {
        float* no = (step == 0) ? obj_A  : outObj;
        float* np = (step == 0) ? pred_A : outPred;
        k_zero<<<(int)(OBJ / 256), 256, 0, stream>>>(msg_obj, (int)OBJ);
        k_pair<<<2048, 256, 0, stream>>>(co, cp, idx, w_e2v, b_e2v, w_v2e, b_v2e,
                                         msg_obj, pred_msg);
        k_gru<<<N_OBJ  / 32, 256, 0, stream>>>(msg_obj,  co, wihB, whhB,
                                               b_ih, b_hh, no, N_OBJ);
        k_gru<<<N_PAIR / 32, 256, 0, stream>>>(pred_msg, cp, wihB, whhB,
                                               b_ih, b_hh, np, N_PAIR);
        co = no; cp = np;
    }
}
